// PureLSTM_39908836114600
// MI455X (gfx1250) — matrix operand order repaired and checked by reading, not compiled, not run on hardware
//
#include <hip/hip_runtime.h>

#define HDIM 32
#define SEQ  512
#define FEAT 4
#define BATCH 8192
#define WAVES_PER_BLOCK 4
#define ROWS_PER_WAVE 16
#define WSTRIDE 40  // halfs per weight row in LDS (80B) -> conflict-free b128 loads

typedef __attribute__((ext_vector_type(16))) _Float16 v16h;
typedef __attribute__((ext_vector_type(8)))  _Float16 v8h;
typedef __attribute__((ext_vector_type(2)))  __fp16   v2fp16;  // cvt_pkrtz result type
typedef __attribute__((ext_vector_type(8)))  float    v8f;
typedef __attribute__((ext_vector_type(4)))  float    v4f;
typedef __attribute__((ext_vector_type(2)))  float    v2f;

// ---- activations: CDNA5 hardware tanh (single TRANS op) ----
#if __has_builtin(__builtin_amdgcn_tanhf)
static __device__ __forceinline__ float tanh_(float x) {
  return __builtin_amdgcn_tanhf(x);
}
static __device__ __forceinline__ float sigm(float x) {
  // sigmoid(x) = 0.5*tanh(x/2) + 0.5  -> v_mul + v_tanh + v_fma
  return __builtin_fmaf(0.5f, __builtin_amdgcn_tanhf(0.5f * x), 0.5f);
}
#else
static __device__ __forceinline__ float tanh_(float x) {
  // folded constant: exp(-2x) = exp2(x * -2*log2(e))
  return 2.0f * __builtin_amdgcn_rcpf(
             1.0f + __builtin_amdgcn_exp2f(x * -2.88539008177792681472f)) -
         1.0f;
}
static __device__ __forceinline__ float sigm(float x) {
  return __builtin_amdgcn_rcpf(
      1.0f + __builtin_amdgcn_exp2f(x * -1.44269504088896340736f));
}
#endif

// A-operand (16x32 f16) tile T of a [128x32] weight matrix held row-major in LDS.
// Per lane: row M = T*16 + (lane%16); lanes<16 take K0-15, lanes>=16 take K16-31.
static __device__ __forceinline__ v16h ldA16(const _Float16* w, int T, int ln, int hi) {
  const _Float16* row = w + (T * 16 + ln) * WSTRIDE + hi * 16;
  union { v16h v; v8h h[2]; } u;
  u.h[0] = *(const v8h*)(row);        // K 0-7   (or 16-23)
  u.h[1] = *(const v8h*)(row + 8);    // K 8-15  (or 24-31)
  return u.v;
}

// C/D-layout bias tile: VGPR r = bias[T*16 + 8*hi + r]  (broadcast across 16 lanes)
static __device__ __forceinline__ v8f ldBias(const float* b, int T, int hi) {
  const float* p = b + T * 16 + 8 * hi;
  union { v8f v; v4f f[2]; } u;
  u.f[0] = *(const v4f*)(p);
  u.f[1] = *(const v4f*)(p + 4);
  return u.v;
}

// gt[0..7] = gate pre-activation D-tiles (units [i|f|g|o], 16 units per tile).
// u-tile 0 -> gt[0,2,4,6], u-tile 1 -> gt[1,3,5,7]. All elementwise in D layout.
static __device__ __forceinline__ void cellUpdate(const v8f* gt, v8f& ca, v8f& cb,
                                                  v8f& hA, v8f& hB) {
#pragma unroll
  for (int r = 0; r < 8; ++r) {
    float cc0 = sigm(gt[2][r]) * ca[r] + sigm(gt[0][r]) * tanh_(gt[4][r]);
    ca[r] = cc0;
    hA[r] = sigm(gt[6][r]) * tanh_(cc0);
    float cc1 = sigm(gt[3][r]) * cb[r] + sigm(gt[1][r]) * tanh_(gt[5][r]);
    cb[r] = cc1;
    hB[r] = sigm(gt[7][r]) * tanh_(cc1);
  }
}

// Convert two f32 D-tiles of h (units 0-15 / 16-31) into the f16 B operand (32x16).
// B layout: lanes<16 hold K=0..15 (halfs 0..15), lanes>=16 hold K=16..31.
// D gives lanes<16 units {0-7, 16-23} and lanes>=16 units {8-15, 24-31}; one
// xor-16 swap of 4 dwords moves the crossing halves - no LDS transpose needed.
// f32->f16 via packed v_cvt_pk_rtz_f16_f32 (2 values/op).
static __device__ __forceinline__ v16h packB(const v8f& hD0, const v8f& hD1, bool low) {
  union CU { v2fp16 h; unsigned u; };
  unsigned lo[4], hh[4];
#pragma unroll
  for (int j = 0; j < 4; ++j) {
    CU a, b;
    a.h = __builtin_amdgcn_cvt_pkrtz(hD0[2 * j], hD0[2 * j + 1]);
    b.h = __builtin_amdgcn_cvt_pkrtz(hD1[2 * j], hD1[2 * j + 1]);
    lo[j] = a.u;
    hh[j] = b.u;
  }
  unsigned recv[4];
#pragma unroll
  for (int j = 0; j < 4; ++j) {
    unsigned send = low ? hh[j] : lo[j];
    recv[j] = __shfl_xor(send, 16, 32);
  }
  union { v16h v; unsigned u[8]; } b;
#pragma unroll
  for (int j = 0; j < 4; ++j) {
    b.u[j]     = low ? lo[j] : recv[j];
    b.u[4 + j] = low ? recv[j] : hh[j];
  }
  return b.v;
}

__global__ __launch_bounds__(WAVES_PER_BLOCK * 32) void lstm_fused_kernel(
    const float* __restrict__ x,
    const float* __restrict__ Wih0, const float* __restrict__ Whh0,
    const float* __restrict__ bih0, const float* __restrict__ bhh0,
    const float* __restrict__ Wih1, const float* __restrict__ Whh1,
    const float* __restrict__ bih1, const float* __restrict__ bhh1,
    const float* __restrict__ fcw, const float* __restrict__ fcb,
    float* __restrict__ out) {
  __shared__ _Float16 sWhh0[128 * WSTRIDE];
  __shared__ _Float16 sWih1[128 * WSTRIDE];
  __shared__ _Float16 sWhh1[128 * WSTRIDE];
  __shared__ float sWih0[128 * FEAT];
  __shared__ float sB0[128];
  __shared__ float sB1[128];
  __shared__ float sFC[HDIM];

  const int tid = threadIdx.x;
  // Cooperative weight preload (f32 -> f16 for the K=32 matrices).
  for (int i = tid; i < 128 * 32; i += WAVES_PER_BLOCK * 32) {
    int r = i >> 5, k = i & 31;
    sWhh0[r * WSTRIDE + k] = (_Float16)Whh0[i];
    sWih1[r * WSTRIDE + k] = (_Float16)Wih1[i];
    sWhh1[r * WSTRIDE + k] = (_Float16)Whh1[i];
  }
  for (int i = tid; i < 128 * FEAT; i += WAVES_PER_BLOCK * 32) sWih0[i] = Wih0[i];
  if (tid < 128) { sB0[tid] = bih0[tid] + bhh0[tid]; sB1[tid] = bih1[tid] + bhh1[tid]; }
  if (tid < HDIM) sFC[tid] = fcw[tid];
  __syncthreads();

  const int wave = tid >> 5;
  const int lane = tid & 31;
  const int ln = lane & 15;
  const int hi = lane >> 4;
  const bool low = (hi == 0);
  const int b0 = (blockIdx.x * WAVES_PER_BLOCK + wave) * ROWS_PER_WAVE;

  // Per-lane x pointer: B operand (4x16 f32) wants lanes<16 -> features {0,1},
  // lanes>=16 -> features {2,3}, one b64 per step.
  const float* xp = x + (size_t)(b0 + ln) * SEQ * FEAT + 2 * hi;

  v16h h0B = {};              // h as f16 B operand, zero-initialized state
  v16h h1B = {};
  v8f c0a = {}, c0b = {};     // c state in D layout (units 0-15 / 16-31)
  v8f c1a = {}, c1b = {};
  v8f h1D0 = {}, h1D1 = {};   // fp32 last h1, kept for the final fc

#pragma unroll 1
  for (int t = 0; t < SEQ; ++t) {
    v2f xB = *(const v2f*)(xp + (size_t)t * FEAT);

    // ----- layer 0: G^T = b + W_ih0 * x^T (fp32 WMMA, K=4) + W_hh0 * h0^T (f16 WMMA, K=32)
    v8f gt[8];
#pragma unroll
    for (int T = 0; T < 8; ++T) {
      v8f acc = ldBias(sB0, T, hi);
      v2f aih = *(const v2f*)(sWih0 + (T * 16 + ln) * FEAT + 2 * hi);
      acc = __builtin_amdgcn_wmma_f32_16x16x4_f32(false, aih, false, xB,
                                                  (short)0, acc, false, false);
      v16h ahh = ldA16(sWhh0, T, ln, hi);
      acc = __builtin_amdgcn_wmma_f32_16x16x32_f16(false, ahh, false, h0B,
                                                   (short)0, acc, false, false);
      gt[T] = acc;
    }
    v8f h0D0, h0D1;
    cellUpdate(gt, c0a, c0b, h0D0, h0D1);
    h0B = packB(h0D0, h0D1, low);

    // ----- layer 1: G^T = b + W_ih1 * h0^T + W_hh1 * h1^T (both f16 WMMA, K=32)
#pragma unroll
    for (int T = 0; T < 8; ++T) {
      v8f acc = ldBias(sB1, T, hi);
      v16h a1 = ldA16(sWih1, T, ln, hi);
      acc = __builtin_amdgcn_wmma_f32_16x16x32_f16(false, a1, false, h0B,
                                                   (short)0, acc, false, false);
      v16h a2 = ldA16(sWhh1, T, ln, hi);
      acc = __builtin_amdgcn_wmma_f32_16x16x32_f16(false, a2, false, h1B,
                                                   (short)0, acc, false, false);
      gt[T] = acc;
    }
    cellUpdate(gt, c1a, c1b, h1D0, h1D1);
    h1B = packB(h1D0, h1D1, low);
  }

  // out[b] = sigmoid(fc_w . h1[:,b] + fc_b). Lane n holds units {0-7,16-23},
  // lane n+16 holds {8-15,24-31}; combine with one xor-16 shuffle.
  float partial = 0.0f;
#pragma unroll
  for (int r = 0; r < 8; ++r)
    partial += sFC[8 * hi + r] * h1D0[r] + sFC[16 + 8 * hi + r] * h1D1[r];
  float other = __shfl_xor(partial, 16, 32);
  float dot = partial + other + fcb[0];
  if (lane < 16) out[b0 + ln] = sigm(dot);
}

extern "C" void kernel_launch(void* const* d_in, const int* in_sizes, int n_in,
                              void* d_out, int out_size, void* d_ws, size_t ws_size,
                              hipStream_t stream) {
  const float* x    = (const float*)d_in[0];
  const float* Wih0 = (const float*)d_in[1];
  const float* Whh0 = (const float*)d_in[2];
  const float* bih0 = (const float*)d_in[3];
  const float* bhh0 = (const float*)d_in[4];
  const float* Wih1 = (const float*)d_in[5];
  const float* Whh1 = (const float*)d_in[6];
  const float* bih1 = (const float*)d_in[7];
  const float* bhh1 = (const float*)d_in[8];
  const float* fcw  = (const float*)d_in[9];
  const float* fcb  = (const float*)d_in[10];
  float* out = (float*)d_out;

  const int rows_per_block = WAVES_PER_BLOCK * ROWS_PER_WAVE;  // 64
  dim3 grid(BATCH / rows_per_block);                           // 128 blocks
  dim3 block(WAVES_PER_BLOCK * 32);                            // 128 threads = 4 waves
  lstm_fused_kernel<<<grid, block, 0, stream>>>(x, Wih0, Whh0, bih0, bhh0,
                                                Wih1, Whh1, bih1, bhh1,
                                                fcw, fcb, out);
}
